// variant1_5970004542120
// MI455X (gfx1250) — compile-verified, matrix operand order checked by reading, not run on hardware
//
#include <hip/hip_runtime.h>

typedef _Float16 f16;
typedef _Float16 v16h __attribute__((ext_vector_type(16)));
typedef _Float16 v8h  __attribute__((ext_vector_type(8)));
typedef float    v8f  __attribute__((ext_vector_type(8)));

union V16U { v16h v; v8h h[2]; };

__device__ inline v8f wmma16(v16h a, v16h b, v8f c) {
  // 8 args: (neg_a, A, neg_b, B, c_mod, C, reuse_a, reuse_b)
  return __builtin_amdgcn_wmma_f32_16x16x32_f16(false, a, false, b, (short)0, c,
                                                false, false);
}

// A-fragment (16x32 f16) built from an f32 row. p points at row + k0 + 8*(lane>>4).
// halves 0..7 <- p[0..7], halves 8..15 <- p[16..23]  (ISA 16-bit A layout)
__device__ inline v16h frag_from_f32(const float* p) {
  v8f lo = { p[0], p[1], p[2], p[3], p[4], p[5], p[6], p[7] };
  v8f hi = { p[16], p[17], p[18], p[19], p[20], p[21], p[22], p[23] };
  V16U u;
  u.h[0] = __builtin_convertvector(lo, v8h);
  u.h[1] = __builtin_convertvector(hi, v8h);
  return u.v;
}

// A-fragment from an f16 LDS tile (row-major, 128 cols). Same two 16B runs.
__device__ inline v16h frag_from_lds(const f16* p) {
  V16U u;
  u.h[0] = *(const v8h*)p;
  u.h[1] = *(const v8h*)(p + 16);
  return u.v;
}

// Pre-packed weight fragment: 16 halves contiguous per lane (2x ds_load_b128).
__device__ inline v16h frag_packed(const f16* p) {
  V16U u;
  const v8h* q = (const v8h*)p;
  u.h[0] = q[0];
  u.h[1] = q[1];
  return u.v;
}

// Order LDS traffic across lanes within a wave (no workgroup barrier needed).
__device__ inline void wave_lds_fence() {
  __builtin_amdgcn_fence(__ATOMIC_ACQ_REL, "wavefront");
  __builtin_amdgcn_wave_barrier();
}

// Pack weight W[K][Ncol] (f32, row-major) into per-fragment lane-major f16 LDS:
// dst[((kt*NT+nt)*32 + lane)*16 + h] = W[kt*32 + k(lane,h)][nt*16 + (lane&15)]
__device__ inline void packW(const float* __restrict__ W, f16* dst, int K,
                             int Ncol, int NT) {
  const int total = (K >> 5) * NT * 32;
  for (int idx = threadIdx.x; idx < total; idx += blockDim.x) {
    const int lane = idx & 31, t = idx >> 5;
    const int kt = t / NT, nt = t - kt * NT;
    const int nn = nt * 16 + (lane & 15);
    f16* d = dst + ((size_t)t * 32 + (size_t)lane) * 16;
#pragma unroll
    for (int h = 0; h < 16; ++h) {
      const int kk = kt * 32 + h + 8 * ((h >> 3) + (lane >> 4));
      d[h] = (f16)W[(size_t)kk * Ncol + nn];
    }
  }
}

__device__ inline float lrelu(float v) { return v > 0.f ? v : 0.2f * v; }

// ---------------------------------------------------------------------------
// Edge pipeline: pe = relu(ea@W1+b1)@W2+b2 ; logit = relu(pe@Wa1+ba1)@Wa2+ba2
// One 16-edge tile per wave; all three GEMMs via v_wmma_f32_16x16x32_f16.
// ---------------------------------------------------------------------------
__global__ void __launch_bounds__(256) k_edge_mlp(
    const float* __restrict__ ea, const float* __restrict__ W1,
    const float* __restrict__ b1, const float* __restrict__ W2,
    const float* __restrict__ b2, const float* __restrict__ Wa1,
    const float* __restrict__ ba1, const float* __restrict__ Wa2,
    const float* __restrict__ ba2, float* __restrict__ pe,
    float* __restrict__ logits, int E) {
  extern __shared__ char smem[];
  f16* w1p = (f16*)smem;       // 1x8 tiles  * 512 halves = 4096
  f16* w2p = w1p + 4096;       // 4x8 tiles  * 512        = 16384
  f16* wa1p = w2p + 16384;     // 4x4 tiles  * 512        = 8192
  f16* stage = wa1p + 8192;    // 8 waves * 16x128 halves = 16384
  float* fb = (float*)(stage + 8 * 2048);
  float* b1s = fb;
  float* b2s = fb + 128;
  float* ba1s = fb + 256;
  float* wa2s = fb + 320;

  packW(W1, w1p, 32, 128, 8);
  packW(W2, w2p, 128, 128, 8);
  packW(Wa1, wa1p, 128, 64, 4);
  for (int i = threadIdx.x; i < 128; i += blockDim.x) {
    b1s[i] = b1[i];
    b2s[i] = b2[i];
  }
  for (int i = threadIdx.x; i < 64; i += blockDim.x) {
    ba1s[i] = ba1[i];
    wa2s[i] = Wa2[i];
  }
  __syncthreads();

  const float ba2v = ba2[0];
  const int lane = threadIdx.x & 31, wave = threadIdx.x >> 5;
  const int lm = lane & 15, hw = lane >> 4;
  f16* st = stage + wave * 2048;
  const int nTiles = (E + 15) >> 4;
  const v8f z = {};

  for (int tile = blockIdx.x * 8 + wave; tile < nTiles; tile += gridDim.x * 8) {
    const int e0 = tile << 4;
    int row = e0 + lm;
    if (row >= E) row = E - 1;

    // GEMM1: h1 = relu(ea[16x32] @ W1[32x128] + b1) -> stage (f16)
    const v16h aG = frag_from_f32(ea + (size_t)row * 32 + 8 * hw);
#pragma unroll
    for (int nt = 0; nt < 8; ++nt) {
      v8f c = {};
      c = wmma16(aG, frag_packed(w1p + nt * 512 + lane * 16), c);
      const float bb = b1s[nt * 16 + lm];
#pragma unroll
      for (int r = 0; r < 8; ++r) {
        float v = c[r] + bb;
        v = fmaxf(v, 0.f);
        st[(r + 8 * hw) * 128 + nt * 16 + lm] = (f16)v;
      }
    }
    wave_lds_fence();

    // GEMM2: pe = h1[16x128] @ W2[128x128] + b2 -> global f32 + stage (f16)
    v8f acc[8];
#pragma unroll
    for (int nt = 0; nt < 8; ++nt) acc[nt] = z;
#pragma unroll
    for (int kt = 0; kt < 4; ++kt) {
      const v16h aF = frag_from_lds(st + lm * 128 + kt * 32 + 8 * hw);
#pragma unroll
      for (int nt = 0; nt < 8; ++nt)
        acc[nt] =
            wmma16(aF, frag_packed(w2p + (kt * 8 + nt) * 512 + lane * 16),
                   acc[nt]);
    }
    wave_lds_fence();
#pragma unroll
    for (int nt = 0; nt < 8; ++nt) {
      const float bb = b2s[nt * 16 + lm];
#pragma unroll
      for (int r = 0; r < 8; ++r) {
        const float v = acc[nt][r] + bb;
        const int er = e0 + r + 8 * hw;
        if (er < E) pe[(size_t)er * 128 + nt * 16 + lm] = v;
        st[(r + 8 * hw) * 128 + nt * 16 + lm] = (f16)v;  // NOT relu'd
      }
    }
    wave_lds_fence();

    // GEMM3: h2 = pe[16x128] @ Wa1[128x64]; logit = relu(h2+ba1) @ Wa2 + ba2
    v8f a3[4];
#pragma unroll
    for (int nt = 0; nt < 4; ++nt) a3[nt] = z;
#pragma unroll
    for (int kt = 0; kt < 4; ++kt) {
      const v16h aF = frag_from_lds(st + lm * 128 + kt * 32 + 8 * hw);
#pragma unroll
      for (int nt = 0; nt < 4; ++nt)
        a3[nt] =
            wmma16(aF, frag_packed(wa1p + (kt * 4 + nt) * 512 + lane * 16),
                   a3[nt]);
    }
#pragma unroll
    for (int r = 0; r < 8; ++r) {
      float s = 0.f;
#pragma unroll
      for (int t = 0; t < 4; ++t) {
        const int n = t * 16 + lm;
        float v = a3[t][r] + ba1s[n];
        v = fmaxf(v, 0.f);
        s += v * wa2s[n];
      }
      s += __shfl_xor(s, 1, 16);
      s += __shfl_xor(s, 2, 16);
      s += __shfl_xor(s, 4, 16);
      s += __shfl_xor(s, 8, 16);
      if (lm == 0) {
        const int er = e0 + r + 8 * hw;
        if (er < E) logits[er] = s + ba2v;
      }
    }
    wave_lds_fence();
  }
}

// ---------------------------------------------------------------------------
// xp = x[N x 128] @ W_gat[128 x 128]   (WMMA, A-fragments straight from HBM)
// ---------------------------------------------------------------------------
__global__ void __launch_bounds__(256) k_xp(const float* __restrict__ x,
                                            const float* __restrict__ Wg,
                                            float* __restrict__ xp, int N) {
  extern __shared__ char smem[];
  f16* wp = (f16*)smem;  // 4x8 tiles * 512 halves = 16384
  packW(Wg, wp, 128, 128, 8);
  __syncthreads();
  const int lane = threadIdx.x & 31, wave = threadIdx.x >> 5;
  const int lm = lane & 15, hw = lane >> 4;
  const int nTiles = (N + 15) >> 4;
  const v8f z = {};
  for (int tile = blockIdx.x * 8 + wave; tile < nTiles; tile += gridDim.x * 8) {
    const int n0 = tile << 4;
    int row = n0 + lm;
    if (row >= N) row = N - 1;
    v8f acc[8];
#pragma unroll
    for (int nt = 0; nt < 8; ++nt) acc[nt] = z;
#pragma unroll
    for (int kt = 0; kt < 4; ++kt) {
      const v16h aF = frag_from_f32(x + (size_t)row * 128 + kt * 32 + 8 * hw);
#pragma unroll
      for (int nt = 0; nt < 8; ++nt)
        acc[nt] = wmma16(aF, frag_packed(wp + (kt * 8 + nt) * 512 + lane * 16),
                         acc[nt]);
    }
#pragma unroll
    for (int nt = 0; nt < 8; ++nt)
#pragma unroll
      for (int r = 0; r < 8; ++r) {
        const int nr = n0 + r + 8 * hw;
        if (nr < N) xp[(size_t)nr * 128 + nt * 16 + lm] = acc[nt][r];
      }
  }
}

// ---------------------------------------------------------------------------
// GAT attention scalars + segment softmax + scatters (VMEM/atomic kernels)
// ---------------------------------------------------------------------------
__global__ void k_attn(const float* __restrict__ xp, const float* __restrict__ aS,
                       const float* __restrict__ aD, float* asrc, float* adst,
                       float* mbuf, int N) {
  const int idx = blockIdx.x * blockDim.x + threadIdx.x;
  if (idx >= N * 4) return;
  const int n = idx >> 2, h = idx & 3;
  const float* p = xp + (size_t)n * 128 + h * 32;
  const float* s = aS + h * 32;
  const float* d = aD + h * 32;
  float ss = 0.f, dd = 0.f;
#pragma unroll 8
  for (int c = 0; c < 32; ++c) {
    const float v = p[c];
    ss += v * s[c];
    dd += v * d[c];
  }
  asrc[idx] = ss;
  adst[idx] = dd;
  mbuf[idx] = lrelu(ss + dd);  // self-loop initializes the segment max
}

__device__ inline void atomicMaxF(float* a, float v) {
  int* ai = (int*)a;
  int cur = __float_as_int(*a);
  while (__int_as_float(cur) < v) {
    const int old = atomicCAS(ai, cur, __float_as_int(v));
    if (old == cur) break;
    cur = old;
  }
}

__global__ void k_edge_max(const int* __restrict__ ei,
                           const float* __restrict__ asrc,
                           const float* __restrict__ adst, float* mbuf, int E) {
  const int idx = blockIdx.x * blockDim.x + threadIdx.x;
  if (idx >= E * 4) return;
  const int e = idx >> 2, h = idx & 3;
  const int s = ei[e], d = ei[E + e];
  atomicMaxF(&mbuf[d * 4 + h], lrelu(asrc[s * 4 + h] + adst[d * 4 + h]));
}

__global__ void k_denom_init(const float* asrc, const float* adst,
                             const float* mbuf, float* den, int N) {
  const int idx = blockIdx.x * blockDim.x + threadIdx.x;
  if (idx >= N * 4) return;
  den[idx] = __expf(lrelu(asrc[idx] + adst[idx]) - mbuf[idx]);
}

__global__ void k_expsum(const int* __restrict__ ei, const float* asrc,
                         const float* adst, const float* mbuf, float* den,
                         float* exb, int E) {
  const int idx = blockIdx.x * blockDim.x + threadIdx.x;
  if (idx >= E * 4) return;
  const int e = idx >> 2, h = idx & 3;
  const int s = ei[e], d = ei[E + e];
  const float v = lrelu(asrc[s * 4 + h] + adst[d * 4 + h]);
  const float ev = __expf(v - mbuf[d * 4 + h]);
  exb[idx] = ev;
  atomicAdd(&den[d * 4 + h], ev);
}

__global__ void k_hinit(const float* asrc, const float* adst, const float* mbuf,
                        const float* den, const float* __restrict__ xp,
                        const float* __restrict__ bg, float* h, int N) {
  const int idx = blockIdx.x * blockDim.x + threadIdx.x;
  if (idx >= N * 128) return;
  const int n = idx >> 7, j = idx & 127, hh = j >> 5;
  const float se =
      __expf(lrelu(asrc[n * 4 + hh] + adst[n * 4 + hh]) - mbuf[n * 4 + hh]);
  h[idx] = bg[j] + (se / den[n * 4 + hh]) * xp[idx];
}

__global__ void k_msg(const int* __restrict__ ei, const float* __restrict__ exb,
                      const float* __restrict__ den,
                      const float* __restrict__ xp, float* h, int E) {
  const int wid = (blockIdx.x * blockDim.x + threadIdx.x) >> 5;
  const int lane = threadIdx.x & 31;
  if (wid >= E) return;
  const int s = ei[wid], d = ei[E + wid];
  const int head = lane >> 3;  // (lane*4)>>5
  const float alpha = exb[wid * 4 + head] / den[d * 4 + head];
  const float4 xv = *(const float4*)(xp + (size_t)s * 128 + lane * 4);
  float* hp = h + (size_t)d * 128 + lane * 4;
  atomicAdd(hp + 0, alpha * xv.x);
  atomicAdd(hp + 1, alpha * xv.y);
  atomicAdd(hp + 2, alpha * xv.z);
  atomicAdd(hp + 3, alpha * xv.w);
}

__global__ void k_elu(float* h, int n) {
  const int idx = blockIdx.x * blockDim.x + threadIdx.x;
  if (idx >= n) return;
  const float v = h[idx];
  h[idx] = v > 0.f ? v : (__expf(v) - 1.f);
}

__global__ void k_agg(const int* __restrict__ ei,
                      const float* __restrict__ logits,
                      const float* __restrict__ scal,
                      const float* __restrict__ pe, float* h, int E) {
  const int wid = (blockIdx.x * blockDim.x + threadIdx.x) >> 5;
  const int lane = threadIdx.x & 31;
  if (wid >= E) return;
  const float w = __expf(logits[wid] - scal[0]) * scal[1];
  const int s = ei[wid];
  const float4 pv = *(const float4*)(pe + (size_t)wid * 128 + lane * 4);
  float* hp = h + (size_t)s * 128 + lane * 4;
  atomicAdd(hp + 0, w * pv.x);
  atomicAdd(hp + 1, w * pv.y);
  atomicAdd(hp + 2, w * pv.z);
  atomicAdd(hp + 3, w * pv.w);
}

// global softmax over E logits: partial/final max, partial/final sum
__global__ void k_rmax(const float* __restrict__ l, int E, float* part) {
  __shared__ float sm[256];
  float m = -3.0e38f;
  for (int i = blockIdx.x * blockDim.x + threadIdx.x; i < E;
       i += gridDim.x * blockDim.x)
    m = fmaxf(m, l[i]);
  sm[threadIdx.x] = m;
  __syncthreads();
  for (int s = 128; s > 0; s >>= 1) {
    if (threadIdx.x < s)
      sm[threadIdx.x] = fmaxf(sm[threadIdx.x], sm[threadIdx.x + s]);
    __syncthreads();
  }
  if (threadIdx.x == 0) part[blockIdx.x] = sm[0];
}

__global__ void k_rmax_fin(const float* part, float* scal, int P) {
  __shared__ float sm[256];
  float m = -3.0e38f;
  for (int i = threadIdx.x; i < P; i += blockDim.x) m = fmaxf(m, part[i]);
  sm[threadIdx.x] = m;
  __syncthreads();
  for (int s = 128; s > 0; s >>= 1) {
    if (threadIdx.x < s)
      sm[threadIdx.x] = fmaxf(sm[threadIdx.x], sm[threadIdx.x + s]);
    __syncthreads();
  }
  if (threadIdx.x == 0) scal[0] = sm[0];
}

__global__ void k_rsum(const float* __restrict__ l, int E, const float* scal,
                       float* part) {
  __shared__ float sm[256];
  const float M = scal[0];
  float s = 0.f;
  for (int i = blockIdx.x * blockDim.x + threadIdx.x; i < E;
       i += gridDim.x * blockDim.x)
    s += __expf(l[i] - M);
  sm[threadIdx.x] = s;
  __syncthreads();
  for (int k = 128; k > 0; k >>= 1) {
    if (threadIdx.x < k) sm[threadIdx.x] += sm[threadIdx.x + k];
    __syncthreads();
  }
  if (threadIdx.x == 0) part[blockIdx.x] = sm[0];
}

__global__ void k_rsum_fin(const float* part, float* scal, int P) {
  __shared__ float sm[256];
  float s = 0.f;
  for (int i = threadIdx.x; i < P; i += blockDim.x) s += part[i];
  sm[threadIdx.x] = s;
  __syncthreads();
  for (int k = 128; k > 0; k >>= 1) {
    if (threadIdx.x < k) sm[threadIdx.x] += sm[threadIdx.x + k];
    __syncthreads();
  }
  if (threadIdx.x == 0) scal[1] = 1.f / sm[0];
}

__global__ void k_zero_pool(float* pooled, float* cnt, int G) {
  for (int i = threadIdx.x; i < G * 128; i += blockDim.x) pooled[i] = 0.f;
  if (threadIdx.x < G) cnt[threadIdx.x] = 0.f;
}

__global__ void k_pool(const int* __restrict__ batch,
                       const float* __restrict__ h, float* pooled, float* cnt,
                       int N) {
  const int wid = (blockIdx.x * blockDim.x + threadIdx.x) >> 5;
  const int lane = threadIdx.x & 31;
  if (wid >= N) return;
  const int g = batch[wid];
  const float4 v = *(const float4*)(h + (size_t)wid * 128 + lane * 4);
  float* pp = pooled + g * 128 + lane * 4;
  atomicAdd(pp + 0, v.x);
  atomicAdd(pp + 1, v.y);
  atomicAdd(pp + 2, v.z);
  atomicAdd(pp + 3, v.w);
  if (lane == 0) atomicAdd(&cnt[g], 1.f);
}

__global__ void k_final(const float* pooled, const float* cnt,
                        const float* __restrict__ Wf, const float* bf,
                        float* out, int G) {
  const int g = threadIdx.x;
  if (g >= G) return;
  float s = 0.f;
  for (int j = 0; j < 128; ++j) s += pooled[g * 128 + j] * Wf[j];
  out[g] = s / fmaxf(cnt[g], 1.f) + bf[0];
}

// ---------------------------------------------------------------------------
extern "C" void kernel_launch(void* const* d_in, const int* in_sizes, int n_in,
                              void* d_out, int out_size, void* d_ws,
                              size_t ws_size, hipStream_t stream) {
  const float* x = (const float*)d_in[0];
  const int* ei = (const int*)d_in[1];
  const float* ea = (const float*)d_in[2];
  const int* bat = (const int*)d_in[3];
  const float* Wg = (const float*)d_in[4];
  const float* aS = (const float*)d_in[5];
  const float* aD = (const float*)d_in[6];
  const float* bg = (const float*)d_in[7];
  const float* W1 = (const float*)d_in[8];
  const float* b1 = (const float*)d_in[9];
  const float* W2 = (const float*)d_in[10];
  const float* b2 = (const float*)d_in[11];
  const float* Wa1 = (const float*)d_in[12];
  const float* ba1 = (const float*)d_in[13];
  const float* Wa2 = (const float*)d_in[14];
  const float* ba2 = (const float*)d_in[15];
  const float* Wf = (const float*)d_in[16];
  const float* bf = (const float*)d_in[17];
  (void)n_in;
  (void)ws_size;

  const int N = in_sizes[0] / 128;
  const int E = in_sizes[2] / 32;
  const int G = out_size;

  float* ws = (float*)d_ws;
  size_t o = 0;
  float* pe = ws + o;     o += (size_t)E * 128;
  float* xp = ws + o;     o += (size_t)N * 128;
  float* logits = ws + o; o += (size_t)E;
  float* exb = ws + o;    o += (size_t)E * 4;
  float* asrc = ws + o;   o += (size_t)N * 4;
  float* adst = ws + o;   o += (size_t)N * 4;
  float* mbuf = ws + o;   o += (size_t)N * 4;
  float* den = ws + o;    o += (size_t)N * 4;
  float* hb = ws + o;     o += (size_t)N * 128;
  float* pooled = ws + o; o += (size_t)G * 128;
  float* cnt = ws + o;    o += (size_t)G;
  float* part = ws + o;   o += 256;
  float* scal = ws + o;   o += 2;

  const size_t smemA = (size_t)(4096 + 16384 + 8192 + 8 * 2048) * sizeof(f16) +
                       384 * sizeof(float);  // ~89.5 KB (fits 320 KB WGP LDS)
  const size_t smemX = (size_t)16384 * sizeof(f16);  // 32 KB

  k_edge_mlp<<<1024, 256, smemA, stream>>>(ea, W1, b1, W2, b2, Wa1, ba1, Wa2,
                                           ba2, pe, logits, E);
  k_xp<<<512, 256, smemX, stream>>>(x, Wg, xp, N);

  const int tN4 = (N * 4 + 255) / 256;
  const int tE4 = (E * 4 + 255) / 256;
  const int tNH = (int)(((size_t)N * 128 + 255) / 256);

  k_attn<<<tN4, 256, 0, stream>>>(xp, aS, aD, asrc, adst, mbuf, N);

  k_rmax<<<256, 256, 0, stream>>>(logits, E, part);
  k_rmax_fin<<<1, 256, 0, stream>>>(part, scal, 256);
  k_rsum<<<256, 256, 0, stream>>>(logits, E, scal, part);
  k_rsum_fin<<<1, 256, 0, stream>>>(part, scal, 256);

  k_edge_max<<<tE4, 256, 0, stream>>>(ei, asrc, adst, mbuf, E);
  k_denom_init<<<tN4, 256, 0, stream>>>(asrc, adst, mbuf, den, N);
  k_expsum<<<tE4, 256, 0, stream>>>(ei, asrc, adst, mbuf, den, exb, E);
  k_hinit<<<tNH, 256, 0, stream>>>(asrc, adst, mbuf, den, xp, bg, hb, N);
  k_msg<<<(E + 7) / 8, 256, 0, stream>>>(ei, exb, den, xp, hb, E);
  k_elu<<<tNH, 256, 0, stream>>>(hb, N * 128);
  k_agg<<<(E + 7) / 8, 256, 0, stream>>>(ei, logits, scal, pe, hb, E);

  k_zero_pool<<<1, 256, 0, stream>>>(pooled, cnt, G);
  k_pool<<<(N + 7) / 8, 256, 0, stream>>>(bat, hb, pooled, cnt, N);
  k_final<<<1, 32, 0, stream>>>(pooled, cnt, Wf, bf, (float*)d_out, G);
}
